// MixtureOfExpert_22703197127088
// MI455X (gfx1250) — matrix operand order repaired and checked by reading, not compiled, not run on hardware
//
#include <hip/hip_runtime.h>
#include <hip/hip_bf16.h>
#include <stdint.h>

// Problem constants: B=8, S=2048 -> T=16384 tokens, D=1024, E=8, top-2.
#define D_DIM 1024
#define T_TOK 16384
#define NE    8
#define BK    32           // one bf16 WMMA K-step
#define BM    128          // rows per block
#define BN    128          // cols per block
#define LDA   40           // A LDS row stride in ushorts (80B: 16B-aligned, conflict-free)
#define NKB   (D_DIM / BK) // 32 K-steps
#define NBF   (D_DIM / 16) // 64 B-fragment columns per matrix

// Optional CDNA5 async global->LDS staging (ASYNCcnt path); falls back to
// register double-buffering when the builtins are absent.
#if defined(__has_builtin)
#if __has_builtin(__builtin_amdgcn_global_load_async_to_lds_b128) && \
    __has_builtin(__builtin_amdgcn_s_wait_asynccnt)
#define USE_ASYNC 1
#endif
#endif
#ifndef USE_ASYNC
#define USE_ASYNC 0
#endif

typedef __attribute__((ext_vector_type(16))) __bf16 v16bf;
typedef __attribute__((ext_vector_type(8)))  __bf16 v8bf;
typedef __attribute__((ext_vector_type(8)))  float  v8f;
typedef __attribute__((__vector_size__(16))) int    v4i;

union FragBF { v16bf v; v8bf h[2]; };

static __device__ __forceinline__ unsigned short f2bf(float f) {
  unsigned int u = __builtin_bit_cast(unsigned int, f);
  u += 0x7fffu + ((u >> 16) & 1u);                 // round-to-nearest-even
  return (unsigned short)(u >> 16);
}
// packed bf16x8 add via vector bf16 (lowers to packed bf16 HW ops)
static __device__ __forceinline__ uint4 add_bf16x8(uint4 a, uint4 b) {
  v8bf x = __builtin_bit_cast(v8bf, a);
  v8bf y = __builtin_bit_cast(v8bf, b);
  v8bf r = x + y;
  return __builtin_bit_cast(uint4, r);
}

#if USE_ASYNC
// Builtin signature (from toolchain diagnostic): first arg is v4i addrspace(1)*.
static __device__ __forceinline__ void async_cp16(const unsigned short* g, unsigned short* l) {
  __builtin_amdgcn_global_load_async_to_lds_b128(
      (__attribute__((address_space(1))) v4i*)g,
      (__attribute__((address_space(3))) v4i*)l, 0, 0);
}
#endif

// ---------------------------------------------------------------------------
// Router: one wave32 per token, float4 vectorized. Fuses X fp32->bf16.
// ---------------------------------------------------------------------------
__global__ __launch_bounds__(256)
void router_kernel(const float* __restrict__ X, const float* __restrict__ Wr,
                   const float* __restrict__ br, unsigned short* __restrict__ Xb,
                   float* __restrict__ wbuf, int* __restrict__ counts,
                   int* __restrict__ lists) {
  const int lane  = threadIdx.x & 31;
  const int wv    = threadIdx.x >> 5;
  const int token = blockIdx.x * 8 + wv;

  const float4* x4 = (const float4*)(X + (size_t)token * D_DIM);
  float acc[NE];
#pragma unroll
  for (int e = 0; e < NE; ++e) acc[e] = 0.f;

  for (int j = lane; j < D_DIM / 4; j += 32) {
    float4 x = x4[j];
    unsigned short o[4] = {f2bf(x.x), f2bf(x.y), f2bf(x.z), f2bf(x.w)};
    *reinterpret_cast<uint2*>(Xb + (size_t)token * D_DIM + j * 4) =
        *reinterpret_cast<const uint2*>(o);
#pragma unroll
    for (int e = 0; e < NE; ++e) {
      float4 w = ((const float4*)(Wr + e * D_DIM))[j];
      acc[e] += x.x * w.x + x.y * w.y + x.z * w.z + x.w * w.w;
    }
  }
#pragma unroll
  for (int e = 0; e < NE; ++e) {
    float v = acc[e];
#pragma unroll
    for (int off = 16; off >= 1; off >>= 1) v += __shfl_xor(v, off, 32);
    acc[e] = v;
  }
  if (lane == 0) {
    float l[NE], p[NE];
    float m = -3.4e38f;
#pragma unroll
    for (int e = 0; e < NE; ++e) { l[e] = acc[e] + br[e]; m = fmaxf(m, l[e]); }
    float s = 0.f;
#pragma unroll
    for (int e = 0; e < NE; ++e) { p[e] = __expf(l[e] - m); s += p[e]; }
    const float inv = 1.f / s;
    int i0 = 0;
#pragma unroll
    for (int e = 1; e < NE; ++e) if (p[e] > p[i0]) i0 = e;
    int i1 = (i0 == 0) ? 1 : 0;
#pragma unroll
    for (int e = 0; e < NE; ++e) if (e != i0 && p[e] > p[i1]) i1 = e;

    wbuf[token * 2 + 0] = p[i0] * inv;
    wbuf[token * 2 + 1] = p[i1] * inv;
    int pos0 = atomicAdd(&counts[i0], 1);
    lists[i0 * T_TOK + pos0] = token * 2 + 0;
    int pos1 = atomicAdd(&counts[i1], 1);
    lists[i1 * T_TOK + pos1] = token * 2 + 1;
  }
}

// ---------------------------------------------------------------------------
// Pre-swizzle weights into WMMA B-fragment order (bf16), half-split layout:
// [kb][nb][half][lane][8] -> conflict-free ds_read_b128 frag loads and
// contiguous 8KB per-block B staging.
// ---------------------------------------------------------------------------
__global__ __launch_bounds__(256)
void swizzle_weights(const float* __restrict__ We, const float* __restrict__ Wo,
                     unsigned short* __restrict__ Websw, unsigned short* __restrict__ Wobsw) {
  const int z = blockIdx.z;
  const float* src = (z < NE) ? (We + (size_t)z * D_DIM * D_DIM) : Wo;
  unsigned short* dst = (z < NE) ? (Websw + (size_t)z * D_DIM * D_DIM) : Wobsw;

  const int g    = blockIdx.x * 256 + threadIdx.x;
  const int lane = g & 31;
  const int half = (g >> 5) & 1;
  const int nb   = (g >> 6) & (NBF - 1);
  const int kb   = g >> 12;

  const int n = nb * 16 + (lane & 15);
  const int k = kb * 32 + half * 16 + (lane >> 4) * 8;
  const float* s = src + (size_t)n * D_DIM + k;
  unsigned short o[8];
#pragma unroll
  for (int i = 0; i < 8; ++i) o[i] = f2bf(s[i]);
  *reinterpret_cast<uint4*>(dst + ((size_t)(kb * NBF + nb) * 512 + half * 256 + lane * 8)) =
      *reinterpret_cast<const uint4*>(o);
}

// ---------------------------------------------------------------------------
// Fragment loaders (ISA 7.12.2 layouts, wave32).
// ---------------------------------------------------------------------------
static __device__ __forceinline__ v16bf load_fragA(const unsigned short* As, int rowBase, int lane) {
  FragBF f;
  const int r  = rowBase + (lane & 15);
  const int kh = (lane >> 4) * 8;
  f.h[0] = *reinterpret_cast<const v8bf*>(As + r * LDA + kh);
  f.h[1] = *reinterpret_cast<const v8bf*>(As + r * LDA + 16 + kh);
  return f.v;
}
static __device__ __forceinline__ v16bf load_fragB(const unsigned short* Bs, int fragLocal, int lane) {
  FragBF f;
  const unsigned short* base = Bs + fragLocal * 512;
  f.h[0] = *reinterpret_cast<const v8bf*>(base + lane * 8);
  f.h[1] = *reinterpret_cast<const v8bf*>(base + 256 + lane * 8);
  return f.v;
}

#define WMMA8(acc, fa0, fa1, fb0, fb1, fb2, fb3)                                                         \
  acc[0][0] = __builtin_amdgcn_wmma_f32_16x16x32_bf16(false, fa0, false, fb0, (short)0, acc[0][0], false, false); \
  acc[0][1] = __builtin_amdgcn_wmma_f32_16x16x32_bf16(false, fa0, false, fb1, (short)0, acc[0][1], false, false); \
  acc[0][2] = __builtin_amdgcn_wmma_f32_16x16x32_bf16(false, fa0, false, fb2, (short)0, acc[0][2], false, false); \
  acc[0][3] = __builtin_amdgcn_wmma_f32_16x16x32_bf16(false, fa0, false, fb3, (short)0, acc[0][3], false, false); \
  acc[1][0] = __builtin_amdgcn_wmma_f32_16x16x32_bf16(false, fa1, false, fb0, (short)0, acc[1][0], false, false); \
  acc[1][1] = __builtin_amdgcn_wmma_f32_16x16x32_bf16(false, fa1, false, fb1, (short)0, acc[1][1], false, false); \
  acc[1][2] = __builtin_amdgcn_wmma_f32_16x16x32_bf16(false, fa1, false, fb2, (short)0, acc[1][2], false, false); \
  acc[1][3] = __builtin_amdgcn_wmma_f32_16x16x32_bf16(false, fa1, false, fb3, (short)0, acc[1][3], false, false)

// ---------------------------------------------------------------------------
// Grouped expert GEMM. Block: 128 gathered rows x 128 cols; 8 waves (4x2);
// wave = 32x64 = 8 WMMA per K-step. Ping-pong LDS, one barrier per K-step,
// async global->LDS staging when available (else register double-buffer).
// ---------------------------------------------------------------------------
__global__ __launch_bounds__(256)
void expert_gemm(const unsigned short* __restrict__ Xb,
                 const unsigned short* __restrict__ Websw,
                 const float* __restrict__ be,
                 const int* __restrict__ counts, const int* __restrict__ lists,
                 const float* __restrict__ wbuf, unsigned short* __restrict__ Cslot) {
  const int e     = blockIdx.z;
  const int cnt   = counts[e];
  const int mBase = blockIdx.y * BM;
  if (mBase >= cnt) return;
  const int nBase = blockIdx.x * BN;

  const int tid  = threadIdx.x;
  const int lane = tid & 31;
  const int wv   = tid >> 5;
  const int wm   = (wv >> 1) * 32;
  const int wn   = (wv & 1) * 64;

  __shared__ unsigned short As[2][BM * LDA];
  __shared__ unsigned short Bs[2][8 * 512];
  __shared__ int      s_tok[BM];
  __shared__ float    s_w[BM];
  __shared__ unsigned s_rowoff[BM];               // (slot*T + token) * D, fits in 25 bits

  for (int i = tid; i < BM; i += 256) {
    int idx   = mBase + i;
    int cidx  = idx < cnt ? idx : cnt - 1;
    int entry = lists[e * T_TOK + cidx];
    int token = entry >> 1;
    s_tok[i]    = token;
    s_w[i]      = wbuf[entry];
    s_rowoff[i] = ((unsigned)((entry & 1) * T_TOK + token)) << 10;
  }
  __syncthreads();

  const int ar0 = tid >> 2, ar1 = 64 + (tid >> 2);
  const int ach = (tid & 3) * 8;
  const unsigned short* aSrc0 = Xb + (size_t)s_tok[ar0] * D_DIM + ach;
  const unsigned short* aSrc1 = Xb + (size_t)s_tok[ar1] * D_DIM + ach;
  const unsigned short* bSrc =
      Websw + (size_t)e * NKB * NBF * 512 + (size_t)(nBase / 16) * 512 + tid * 8;

  v8f acc[2][4] = {};

  // Prologue: stage K-step 0 into buffer 0
#if USE_ASYNC
  async_cp16(aSrc0, &As[0][ar0 * LDA + ach]);
  async_cp16(aSrc1, &As[0][ar1 * LDA + ach]);
  async_cp16(bSrc,        &Bs[0][tid * 8]);
  async_cp16(bSrc + 2048, &Bs[0][tid * 8 + 2048]);
  __builtin_amdgcn_s_wait_asynccnt(0);
#else
  uint4 ra0 = *reinterpret_cast<const uint4*>(aSrc0);
  uint4 ra1 = *reinterpret_cast<const uint4*>(aSrc1);
  uint4 rb0 = *reinterpret_cast<const uint4*>(bSrc);
  uint4 rb1 = *reinterpret_cast<const uint4*>(bSrc + 2048);
  *reinterpret_cast<uint4*>(&As[0][ar0 * LDA + ach]) = ra0;
  *reinterpret_cast<uint4*>(&As[0][ar1 * LDA + ach]) = ra1;
  *reinterpret_cast<uint4*>(&Bs[0][tid * 8])        = rb0;
  *reinterpret_cast<uint4*>(&Bs[0][tid * 8 + 2048]) = rb1;
#endif
  __syncthreads();

  int cur = 0;
  for (int kb = 0; kb < NKB; ++kb) {
    const bool hasNext = (kb + 1) < NKB;
    if (hasNext) {
      aSrc0 += BK; aSrc1 += BK; bSrc += (size_t)NBF * 512;
#if USE_ASYNC
      async_cp16(aSrc0, &As[cur ^ 1][ar0 * LDA + ach]);
      async_cp16(aSrc1, &As[cur ^ 1][ar1 * LDA + ach]);
      async_cp16(bSrc,        &Bs[cur ^ 1][tid * 8]);
      async_cp16(bSrc + 2048, &Bs[cur ^ 1][tid * 8 + 2048]);
#else
      ra0 = *reinterpret_cast<const uint4*>(aSrc0);
      ra1 = *reinterpret_cast<const uint4*>(aSrc1);
      rb0 = *reinterpret_cast<const uint4*>(bSrc);
      rb1 = *reinterpret_cast<const uint4*>(bSrc + 2048);
#endif
      if (kb + 2 < NKB) {
        __builtin_prefetch(aSrc0 + BK, 0, 1);       // global_prefetch_b8
        __builtin_prefetch(bSrc + (size_t)NBF * 512, 0, 1);
      }
    }

    v16bf fa0 = load_fragA(As[cur], wm, lane);
    v16bf fa1 = load_fragA(As[cur], wm + 16, lane);
    v16bf fb0 = load_fragB(Bs[cur], (wn >> 4) + 0, lane);
    v16bf fb1 = load_fragB(Bs[cur], (wn >> 4) + 1, lane);
    v16bf fb2 = load_fragB(Bs[cur], (wn >> 4) + 2, lane);
    v16bf fb3 = load_fragB(Bs[cur], (wn >> 4) + 3, lane);
    WMMA8(acc, fa0, fa1, fb0, fb1, fb2, fb3);

    if (!hasNext) break;
#if USE_ASYNC
    __builtin_amdgcn_s_wait_asynccnt(0);
#else
    *reinterpret_cast<uint4*>(&As[cur ^ 1][ar0 * LDA + ach]) = ra0;
    *reinterpret_cast<uint4*>(&As[cur ^ 1][ar1 * LDA + ach]) = ra1;
    *reinterpret_cast<uint4*>(&Bs[cur ^ 1][tid * 8])        = rb0;
    *reinterpret_cast<uint4*>(&Bs[cur ^ 1][tid * 8 + 2048]) = rb1;
#endif
    __syncthreads();
    cur ^= 1;
  }

  // Epilogue: precomputed uint row offsets, no per-element 64-bit math
  const int mOff = (lane >> 4) * 8;
  const int nIdx = lane & 15;
#pragma unroll
  for (int s = 0; s < 2; ++s) {
    unsigned ro[8]; float w8[8]; bool ok[8];
#pragma unroll
    for (int r = 0; r < 8; ++r) {
      const int lr = wm + s * 16 + mOff + r;
      ro[r] = s_rowoff[lr];
      w8[r] = s_w[lr];
      ok[r] = (mBase + lr) < cnt;
    }
#pragma unroll
    for (int j = 0; j < 4; ++j) {
      const unsigned colG = (unsigned)(nBase + wn + j * 16 + nIdx);
      const float bias = be[e * D_DIM + colG];
#pragma unroll
      for (int r = 0; r < 8; ++r)
        if (ok[r]) Cslot[ro[r] + colG] = f2bf((acc[s][j][r] + bias) * w8[r]);
    }
  }
}

// ---------------------------------------------------------------------------
// Output projection with fused slot combine:
//   out[t][h] = (Cslot0[t] + Cslot1[t]) . Wob[:][h] + bo[h]   (fp32 out)
// ---------------------------------------------------------------------------
__global__ __launch_bounds__(256)
void proj_gemm(const unsigned short* __restrict__ C0, const unsigned short* __restrict__ C1,
               const unsigned short* __restrict__ Wobsw,
               const float* __restrict__ bo, float* __restrict__ out) {
  const int mBase = blockIdx.y * BM;
  const int nBase = blockIdx.x * BN;
  const int tid  = threadIdx.x;
  const int lane = tid & 31;
  const int wv   = tid >> 5;
  const int wm   = (wv >> 1) * 32;
  const int wn   = (wv & 1) * 64;

  __shared__ unsigned short As[2][BM * LDA];
  __shared__ unsigned short Bs[2][8 * 512];

  const int ar0 = tid >> 2, ar1 = 64 + (tid >> 2);
  const int ach = (tid & 3) * 8;
  size_t aOff0 = (size_t)(mBase + ar0) * D_DIM + ach;
  size_t aOff1 = (size_t)(mBase + ar1) * D_DIM + ach;
  const unsigned short* bSrc = Wobsw + (size_t)(nBase / 16) * 512 + tid * 8;

  v8f acc[2][4] = {};

  uint4 ra0 = add_bf16x8(*reinterpret_cast<const uint4*>(C0 + aOff0),
                         *reinterpret_cast<const uint4*>(C1 + aOff0));
  uint4 ra1 = add_bf16x8(*reinterpret_cast<const uint4*>(C0 + aOff1),
                         *reinterpret_cast<const uint4*>(C1 + aOff1));
  *reinterpret_cast<uint4*>(&As[0][ar0 * LDA + ach]) = ra0;
  *reinterpret_cast<uint4*>(&As[0][ar1 * LDA + ach]) = ra1;
#if USE_ASYNC
  async_cp16(bSrc,        &Bs[0][tid * 8]);
  async_cp16(bSrc + 2048, &Bs[0][tid * 8 + 2048]);
  __builtin_amdgcn_s_wait_asynccnt(0);
#else
  uint4 rb0 = *reinterpret_cast<const uint4*>(bSrc);
  uint4 rb1 = *reinterpret_cast<const uint4*>(bSrc + 2048);
  *reinterpret_cast<uint4*>(&Bs[0][tid * 8])        = rb0;
  *reinterpret_cast<uint4*>(&Bs[0][tid * 8 + 2048]) = rb1;
#endif
  __syncthreads();

  int cur = 0;
  for (int kb = 0; kb < NKB; ++kb) {
    const bool hasNext = (kb + 1) < NKB;
    if (hasNext) {
      aOff0 += BK; aOff1 += BK; bSrc += (size_t)NBF * 512;
#if USE_ASYNC
      async_cp16(bSrc,        &Bs[cur ^ 1][tid * 8]);
      async_cp16(bSrc + 2048, &Bs[cur ^ 1][tid * 8 + 2048]);
#else
      rb0 = *reinterpret_cast<const uint4*>(bSrc);
      rb1 = *reinterpret_cast<const uint4*>(bSrc + 2048);
#endif
      ra0 = add_bf16x8(*reinterpret_cast<const uint4*>(C0 + aOff0),
                       *reinterpret_cast<const uint4*>(C1 + aOff0));
      ra1 = add_bf16x8(*reinterpret_cast<const uint4*>(C0 + aOff1),
                       *reinterpret_cast<const uint4*>(C1 + aOff1));
      if (kb + 2 < NKB) {
        __builtin_prefetch(C0 + aOff0 + BK, 0, 1);
        __builtin_prefetch(bSrc + (size_t)NBF * 512, 0, 1);
      }
    }

    v16bf fa0 = load_fragA(As[cur], wm, lane);
    v16bf fa1 = load_fragA(As[cur], wm + 16, lane);
    v16bf fb0 = load_fragB(Bs[cur], (wn >> 4) + 0, lane);
    v16bf fb1 = load_fragB(Bs[cur], (wn >> 4) + 1, lane);
    v16bf fb2 = load_fragB(Bs[cur], (wn >> 4) + 2, lane);
    v16bf fb3 = load_fragB(Bs[cur], (wn >> 4) + 3, lane);
    WMMA8(acc, fa0, fa1, fb0, fb1, fb2, fb3);

    if (!hasNext) break;
    *reinterpret_cast<uint4*>(&As[cur ^ 1][ar0 * LDA + ach]) = ra0;
    *reinterpret_cast<uint4*>(&As[cur ^ 1][ar1 * LDA + ach]) = ra1;
#if USE_ASYNC
    __builtin_amdgcn_s_wait_asynccnt(0);
#else
    *reinterpret_cast<uint4*>(&Bs[cur ^ 1][tid * 8])        = rb0;
    *reinterpret_cast<uint4*>(&Bs[cur ^ 1][tid * 8 + 2048]) = rb1;
#endif
    __syncthreads();
    cur ^= 1;
  }

  const int mOff = (lane >> 4) * 8;
  const int nIdx = lane & 15;
#pragma unroll
  for (int s = 0; s < 2; ++s) {
#pragma unroll
    for (int j = 0; j < 4; ++j) {
      const int colG = nBase + wn + j * 16 + nIdx;
      const float bias = bo[colG];
#pragma unroll
      for (int r = 0; r < 8; ++r) {
        const int row = mBase + wm + s * 16 + mOff + r;
        out[(size_t)row * D_DIM + colG] = acc[s][j][r] + bias;
      }
    }
  }
}

// ---------------------------------------------------------------------------
extern "C" void kernel_launch(void* const* d_in, const int* in_sizes, int n_in,
                              void* d_out, int out_size, void* d_ws, size_t ws_size,
                              hipStream_t stream) {
  const float* X  = (const float*)d_in[0];
  const float* We = (const float*)d_in[1];
  const float* be = (const float*)d_in[2];
  const float* Wr = (const float*)d_in[3];
  const float* br = (const float*)d_in[4];
  const float* Wo = (const float*)d_in[5];
  const float* bo = (const float*)d_in[6];
  float* out = (float*)d_out;

  char* ws = (char*)d_ws;
  size_t off = 0;
  auto take = [&](size_t bytes) -> char* {
    char* p = ws + off;
    off += (bytes + 255) & ~(size_t)255;
    return p;
  };
  unsigned short* Xb     = (unsigned short*)take((size_t)T_TOK * D_DIM * 2);
  unsigned short* Websw  = (unsigned short*)take((size_t)NE * D_DIM * D_DIM * 2);
  unsigned short* Wobsw  = (unsigned short*)take((size_t)D_DIM * D_DIM * 2);
  float*          wbuf   = (float*)take((size_t)T_TOK * 2 * sizeof(float));
  int*            lists  = (int*)take((size_t)NE * T_TOK * sizeof(int));
  int*            counts = (int*)take(256);
  unsigned short* Cslot  = (unsigned short*)take((size_t)2 * T_TOK * D_DIM * 2);
  unsigned short* C1     = Cslot + (size_t)T_TOK * D_DIM;

  (void)hipMemsetAsync(counts, 0, 256, stream);   // fresh counts every call (graph-safe)

  swizzle_weights<<<dim3((NKB * NBF * 2 * 32) / 256, 1, NE + 1), 256, 0, stream>>>(
      We, Wo, Websw, Wobsw);
  router_kernel<<<T_TOK / 8, 256, 0, stream>>>(X, Wr, br, Xb, wbuf, counts, lists);
  expert_gemm<<<dim3(D_DIM / BN, T_TOK / BM, NE), 256, 0, stream>>>(
      Xb, Websw, be, counts, lists, wbuf, Cslot);
  proj_gemm<<<dim3(D_DIM / BN, T_TOK / BM), 256, 0, stream>>>(
      Cslot, C1, Wobsw, bo, out);
}
